// StereoRefiner_13838384628364
// MI455X (gfx1250) — matrix operand order repaired and checked by reading, not compiled, not run on hardware
//
#include <hip/hip_runtime.h>
#include <math.h>

// ---------------------------------------------------------------------------
// Types / constants
// ---------------------------------------------------------------------------
typedef __attribute__((ext_vector_type(16))) __bf16 v16bf;
typedef __attribute__((ext_vector_type(8)))  float  v8f;

#define DEVFN static __device__ __forceinline__

static constexpr int H_  = 320;
static constexpr int W_  = 960;
static constexpr int P_  = H_ * W_;     // 307200 pixels
static constexpr int TX_ = W_ / 16;     // 60 pixel-tiles per row
static constexpr int NT_ = H_ * TX_;    // 19200 tiles

DEVFN float bf2f(unsigned short h) { return __uint_as_float(((unsigned int)h) << 16); }
DEVFN unsigned short f2bf(float f) {
  unsigned int u = __float_as_uint(f);
  u += 0x7fffu + ((u >> 16) & 1u);                 // round-to-nearest-even
  return (unsigned short)(u >> 16);
}

struct alignas(16) U4 { unsigned int x[4]; };

DEVFN v16bf ld_frag(const unsigned short* p) {    // 16 bf16 = 32B, 16B-aligned
  union { U4 q[2]; v16bf v; } u;
  u.q[0] = *(const U4*)p;
  u.q[1] = *(const U4*)(p + 8);
  return u.v;
}
DEVFN v16bf zero_frag() {
  union { unsigned int w[8]; v16bf v; } u;
#pragma unroll
  for (int j = 0; j < 8; ++j) u.w[j] = 0u;
  return u.v;
}

// ---------------------------------------------------------------------------
// Weight packing: OIHW fp32 -> bf16 in exact WMMA A-fragment lane layout.
// K ordering: k = tap*CINP + c, blocked by 32.  Optionally stacks two weight
// tensors along the output-channel axis (GRU z|r, logits|delta).
// Packed A (this code's k order, matched by B loads): lane L holds row M = L%16; element i -> K = i+16*(L>>4)
// ---------------------------------------------------------------------------
__global__ void pack_weights(const float* __restrict__ w1, int O1,
                             const float* __restrict__ w2, int O2,
                             int I, int CINP, int KS,
                             unsigned short* __restrict__ wp)
{
  const int O   = O1 + O2;
  const int MBv = (O + 15) / 16;
  const int KB  = KS * KS * CINP / 32;
  const size_t total = (size_t)KB * MBv * 512;
  size_t idx = (size_t)blockIdx.x * blockDim.x + threadIdx.x;
  if (idx >= total) return;
  const int i16  = (int)(idx & 15);
  const int lane = (int)((idx >> 4) & 31);
  const size_t t = idx >> 9;
  const int mb   = (int)(t % MBv);
  const int kb   = (int)(t / MBv);
  const int half = lane >> 4;
  const int m    = mb * 16 + (lane & 15);
  const int kk   = i16 + 16 * half;
  const int k    = kb * 32 + kk;
  const int tap  = k / CINP;
  const int c    = k % CINP;
  float v = 0.f;
  if (c < I && m < O) {
    const int ky = tap / KS, kx = tap % KS;
    v = (m < O1) ? w1[(((size_t)m * I + c) * KS + ky) * KS + kx]
                 : w2[(((size_t)(m - O1) * I + c) * KS + ky) * KS + kx];
  }
  wp[idx] = f2bf(v);
}

// ---------------------------------------------------------------------------
// Implicit-GEMM conv with MW x NW register blocking:
// one wave = (MW * 16 outch) x (NW * 16 pixels).  Per K-step it loads
// MW A-fragments + NW B-fragments and issues MW*NW WMMAs -> 16 FLOP/byte of
// L2 traffic at MW=NW=2 (vs 8 unblocked), which matters because the conv
// stack is L2-bandwidth-limited on MI455X, not HBM-limited (all activations
// fit in the 192 MB L2).
// in : NHWC bf16, CIN channels (CIN % 32 == 0)
// out: NHWC bf16, COUT channels (COUT may be unpadded; masked store)
// epi: 0 none, 1 +bias, 2 +bias SiLU, 3 +bias sigmoid, 4 +bias tanh
// ---------------------------------------------------------------------------
template<int CIN, int COUT, int KS, int MW, int NW>
__global__ __launch_bounds__(256)
void conv_wmma(const unsigned short* __restrict__ in,
               const unsigned short* __restrict__ wp,
               const float* __restrict__ b1, const float* __restrict__ b2, int o1,
               unsigned short* __restrict__ out, int dil, int epi)
{
  static_assert(CIN % 32 == 0, "CIN must be multiple of 32");
  constexpr int MB = (COUT + 15) / 16;
  static_assert(MB % MW == 0, "MW must divide MB");
  static_assert(TX_ % NW == 0, "NW must divide TX_");
  constexpr int MG  = MB / MW;
  constexpr int CB  = CIN / 32;
  constexpr int TXG = TX_ / NW;

  const int wid = blockIdx.x * (blockDim.x >> 5) + (threadIdx.x >> 5);
  if (wid >= H_ * TXG * MG) return;              // wave-uniform
  const int lane = threadIdx.x & 31;
  const int mg   = wid % MG;
  const int tile = wid / MG;
  const int y    = tile / TXG;
  const int x0   = (tile % TXG) * (16 * NW);
  const int half = lane >> 4;
  const int lc   = lane & 15;

  v8f acc[MW][NW];
#pragma unroll
  for (int mw = 0; mw < MW; ++mw)
#pragma unroll
    for (int nw = 0; nw < NW; ++nw)
#pragma unroll
      for (int i = 0; i < 8; ++i) acc[mw][nw][i] = 0.f;

#pragma unroll
  for (int tap = 0; tap < KS * KS; ++tap) {
    const int dy = (tap / KS - (KS / 2)) * dil;
    const int dx = (tap % KS - (KS / 2)) * dil;
    const int yy = y + dy;
    if (KS > 1 && (yy < 0 || yy >= H_)) continue;   // wave-uniform skip

    bool ok[NW];
    const unsigned short* bp[NW];
#pragma unroll
    for (int nw = 0; nw < NW; ++nw) {
      const int xx = x0 + nw * 16 + lc + dx;
      ok[nw] = (KS == 1) || ((unsigned)xx < (unsigned)W_);
      bp[nw] = in + ((size_t)yy * W_ + xx) * CIN + 16 * half;
    }

#pragma unroll
    for (int cb = 0; cb < CB; ++cb) {
      v16bf Bf[NW];
#pragma unroll
      for (int nw = 0; nw < NW; ++nw)
        Bf[nw] = ok[nw] ? ld_frag(bp[nw] + cb * 32) : zero_frag();
#pragma unroll
      for (int mw = 0; mw < MW; ++mw) {
        const int mb = mg * MW + mw;
        v16bf A = ld_frag(wp + (((size_t)(tap * CB + cb) * MB + mb) * 32 + lane) * 16);
#pragma unroll
        for (int nw = 0; nw < NW; ++nw)
          acc[mw][nw] = __builtin_amdgcn_wmma_f32_16x16x32_bf16(
              false, A, false, Bf[nw], (short)0, acc[mw][nw], false, false);
      }
    }
  }

#pragma unroll
  for (int mw = 0; mw < MW; ++mw) {
#pragma unroll
    for (int nw = 0; nw < NW; ++nw) {
      const size_t pix = (size_t)y * W_ + x0 + nw * 16 + lc;
#pragma unroll
      for (int r = 0; r < 8; ++r) {
        const int m = (mg * MW + mw) * 16 + r + 8 * half;  // C/D: M = r + 8*half
        if (m < COUT) {
          float v = acc[mw][nw][r];
          if (epi >= 1) {
            float bb = 0.f;
            if (m < o1) { if (b1) bb = b1[m]; }
            else        { if (b2) bb = b2[m - o1]; }
            v += bb;
          }
          if      (epi == 2) v = v / (1.f + __expf(-v));    // SiLU
          else if (epi == 3) v = 1.f / (1.f + __expf(-v));  // sigmoid
          else if (epi == 4) v = tanhf(v);
          out[pix * COUT + m] = f2bf(v);
        }
      }
    }
  }
}

// ---------------------------------------------------------------------------
// Elementwise / norm / misc kernels
// ---------------------------------------------------------------------------
__global__ void pad_image32(const float* __restrict__ img, unsigned short* __restrict__ out)
{
  size_t idx = (size_t)blockIdx.x * blockDim.x + threadIdx.x;
  if (idx >= (size_t)P_ * 32) return;
  const int c = (int)(idx & 31);
  const size_t p = idx >> 5;
  out[idx] = f2bf(c < 3 ? img[(size_t)c * P_ + p] : 0.f);
}

__global__ void zero_f32(float* p, int n)
{
  int i = blockIdx.x * blockDim.x + threadIdx.x;
  if (i < n) p[i] = 0.f;
}

__global__ __launch_bounds__(256)
void chan_stats64(const unsigned short* __restrict__ x, float* __restrict__ stats)
{
  __shared__ float ss[256], sq[256];
  const int c = threadIdx.x & 63;
  const int g = threadIdx.x >> 6;
  float s = 0.f, q = 0.f;
  for (size_t p = (size_t)blockIdx.x * 4 + g; p < (size_t)P_; p += (size_t)gridDim.x * 4) {
    float v = bf2f(x[p * 64 + c]);
    s += v; q += v * v;
  }
  ss[threadIdx.x] = s; sq[threadIdx.x] = q;
  __syncthreads();
  if (g == 0) {
    s = ss[c] + ss[c + 64] + ss[c + 128] + ss[c + 192];
    q = sq[c] + sq[c + 64] + sq[c + 128] + sq[c + 192];
    atomicAdd(&stats[c], s);
    atomicAdd(&stats[64 + c], q);
  }
}

// DomainNorm (instance norm -> per-pixel L2 -> affine) fused with SiLU, in place.
__global__ void dnorm_silu64(unsigned short* __restrict__ x, const float* __restrict__ stats,
                             const float* __restrict__ w, const float* __restrict__ b)
{
  size_t p = (size_t)blockIdx.x * blockDim.x + threadIdx.x;
  if (p >= (size_t)P_) return;
  unsigned short* xp = x + p * 64;
  const float invP = 1.f / (float)P_;
  float ssq = 0.f;
#pragma unroll 8
  for (int c = 0; c < 64; ++c) {
    float mean = stats[c] * invP;
    float var  = stats[64 + c] * invP - mean * mean;
    float rstd = rsqrtf(var + 1e-5f);
    float v = (bf2f(xp[c]) - mean) * rstd;
    ssq += v * v;
  }
  const float rn = 1.f / fmaxf(sqrtf(ssq), 1e-12f);
#pragma unroll 8
  for (int c = 0; c < 64; ++c) {
    float mean = stats[c] * invP;
    float var  = stats[64 + c] * invP - mean * mean;
    float rstd = rsqrtf(var + 1e-5f);
    float v = (bf2f(xp[c]) - mean) * rstd * rn * w[c] + b[c];
    v = v / (1.f + __expf(-v));
    xp[c] = f2bf(v);
  }
}

__global__ void l2norm64(unsigned short* __restrict__ x)
{
  size_t p = (size_t)blockIdx.x * blockDim.x + threadIdx.x;
  if (p >= (size_t)P_) return;
  unsigned short* xp = x + p * 64;
  float ssq = 0.f;
#pragma unroll 8
  for (int c = 0; c < 64; ++c) { float v = bf2f(xp[c]); ssq += v * v; }
  const float rn = 1.f / (sqrtf(ssq) + 1e-6f);
#pragma unroll 8
  for (int c = 0; c < 64; ++c) xp[c] = f2bf(bf2f(xp[c]) * rn);
}

// align_corners bilinear 80x240 -> 320x960, NCHW fp32 -> NHWC bf16 (32ch)
__global__ void resize_fq(const float* __restrict__ fq, unsigned short* __restrict__ out)
{
  size_t idx = (size_t)blockIdx.x * blockDim.x + threadIdx.x;
  if (idx >= (size_t)P_ * 32) return;
  const int c = (int)(idx & 31);
  const size_t p = idx >> 5;
  const int y = (int)(p / W_), x = (int)(p % W_);
  const float fy = y * (79.f / 319.f);
  const float fx = x * (239.f / 959.f);
  int y0 = (int)floorf(fy); float wy = fy - (float)y0; int y1 = y0 + 1 < 80  ? y0 + 1 : 79;
  int x0 = (int)floorf(fx); float wx = fx - (float)x0; int x1 = x0 + 1 < 240 ? x0 + 1 : 239;
  const float* base = fq + (size_t)c * (80 * 240);
  float v00 = base[y0 * 240 + x0], v01 = base[y0 * 240 + x1];
  float v10 = base[y1 * 240 + x0], v11 = base[y1 * 240 + x1];
  float v = (v00 * (1.f - wx) + v01 * wx) * (1.f - wy) + (v10 * (1.f - wx) + v11 * wx) * wy;
  out[idx] = f2bf(v);
}

__global__ void copy_ch(const unsigned short* __restrict__ src, int Cs,
                        unsigned short* __restrict__ dst, int Cd, int off, int C)
{
  size_t idx = (size_t)blockIdx.x * blockDim.x + threadIdx.x;
  if (idx >= (size_t)P_ * C) return;
  const int c = (int)(idx % C);
  const size_t p = idx / C;
  dst[p * Cd + off + c] = src[p * Cs + c];
}

__global__ void zero_ch(unsigned short* __restrict__ dst, int Cd, int off, int C)
{
  size_t idx = (size_t)blockIdx.x * blockDim.x + threadIdx.x;
  if (idx >= (size_t)P_ * C) return;
  const int c = (int)(idx % C);
  const size_t p = idx / C;
  dst[p * Cd + off + c] = 0;
}

// 9-tap horizontal correlation (grid_sample border/bilinear reduction).
// Writes cost channels 0..8 of the 96-channel concat buffer.
__global__ void local_corr9(const unsigned short* __restrict__ FL,
                            const unsigned short* __restrict__ FR,
                            const float* __restrict__ d0,
                            unsigned short* __restrict__ G96)
{
  size_t p = (size_t)blockIdx.x * blockDim.x + threadIdx.x;
  if (p >= (size_t)P_) return;
  const int y = (int)(p / W_), x = (int)(p % W_);
  const float dv = d0[p];
  const unsigned short* fl   = FL + p * 64;
  const unsigned short* frow = FR + (size_t)y * W_ * 64;
#pragma unroll
  for (int dd = 0; dd < 9; ++dd) {
    float ix  = (float)x - (dv + (float)(dd - 4));
    float x0f = floorf(ix);
    float fr  = ix - x0f;
    int i0 = (int)x0f;     i0 = i0 < 0 ? 0 : (i0 > W_ - 1 ? W_ - 1 : i0);
    int i1 = (int)x0f + 1; i1 = i1 < 0 ? 0 : (i1 > W_ - 1 ? W_ - 1 : i1);
    const unsigned short* a = frow + (size_t)i0 * 64;
    const unsigned short* b = frow + (size_t)i1 * 64;
    float s = 0.f;
#pragma unroll 8
    for (int c = 0; c < 64; ++c)
      s += bf2f(fl[c]) * (bf2f(a[c]) * (1.f - fr) + bf2f(b[c]) * fr);
    G96[p * 96 + dd] = f2bf(s);
  }
}

__global__ void ew_rh(const unsigned short* __restrict__ zr,
                      const unsigned short* __restrict__ h,
                      unsigned short* __restrict__ xh)
{
  size_t idx = (size_t)blockIdx.x * blockDim.x + threadIdx.x;
  if (idx >= (size_t)P_ * 96) return;
  const int c = (int)(idx % 96);
  const size_t p = idx / 96;
  xh[p * 192 + 96 + c] = f2bf(bf2f(zr[p * 192 + 96 + c]) * bf2f(h[p * 96 + c]));
}

__global__ void ew_hupd(const unsigned short* __restrict__ zr,
                        const unsigned short* __restrict__ q,
                        unsigned short* __restrict__ h)
{
  size_t idx = (size_t)blockIdx.x * blockDim.x + threadIdx.x;
  if (idx >= (size_t)P_ * 96) return;
  const int c = (int)(idx % 96);
  const size_t p = idx / 96;
  float z  = bf2f(zr[p * 192 + c]);
  float hv = bf2f(h[p * 96 + c]);
  float qv = bf2f(q[p * 96 + c]);
  h[p * 96 + c] = f2bf((1.f - z) * hv + z * qv);
}

// softmax expectation over 9 logits + delta + d0
__global__ void final_out(const unsigned short* __restrict__ ld,
                          const float* __restrict__ d0, float* __restrict__ out)
{
  size_t p = (size_t)blockIdx.x * blockDim.x + threadIdx.x;
  if (p >= (size_t)P_) return;
  float l[10];
#pragma unroll
  for (int i = 0; i < 10; ++i) l[i] = bf2f(ld[p * 10 + i]);
  float mx = l[0];
#pragma unroll
  for (int i = 1; i < 9; ++i) mx = fmaxf(mx, l[i]);
  float se = 0.f, ex = 0.f;
#pragma unroll
  for (int i = 0; i < 9; ++i) {
    float e = __expf(l[i] - mx);
    se += e; ex += e * (float)(i - 4);
  }
  out[p] = d0[p] + ex / se + l[9];
}

// ---------------------------------------------------------------------------
// Host orchestration
// ---------------------------------------------------------------------------
static inline unsigned blocks1d(size_t n) { return (unsigned)((n + 255) / 256); }

// waves = H_ * (TX_/NW) * (MB/MW); 8 waves per 256-thread block
#define CONV(CIN, COUT, KS, MW, NW, IN, WP, B1, B2, O1, OUT, DIL, EPI)                  \
  conv_wmma<CIN, COUT, KS, MW, NW>                                                      \
      <<<(unsigned)((H_ * (TX_ / (NW)) * ((((COUT) + 15) / 16) / (MW)) + 7) / 8), 256,  \
         0, stream>>>(IN, WP, B1, B2, O1, OUT, DIL, EPI)

extern "C" void kernel_launch(void* const* d_in, const int* in_sizes, int n_in,
                              void* d_out, int out_size, void* d_ws, size_t ws_size,
                              hipStream_t stream)
{
  (void)n_in; (void)out_size; (void)ws_size;
  // Input-order detection: JAX tree_flatten (sorted keys) puts Fq_left (614400)
  // first; plain insertion order puts left_img (921600) first.
  const bool sorted = (in_sizes[0] == 614400);
  auto g = [&](int si, int ii) -> const float* {
    return (const float*)d_in[sorted ? si : ii];
  };
  const float* left   = g(3, 0);
  const float* right  = g(38, 1);
  const float* d0     = g(2, 2);
  const float* fqL    = g(0, 3);
  const float* conv0w = g(15, 5),  *conv1w = g(16, 8),  *conv2w = g(17, 11), *conv3w = g(18, 14);
  const float* dn0w = g(20, 6),  *dn0b = g(19, 7);
  const float* dn1w = g(22, 9),  *dn1b = g(21, 10);
  const float* dn2w = g(24, 12), *dn2b = g(23, 13);
  const float* dn3w = g(26, 15), *dn3b = g(25, 16);
  const float* qprojw = g(34, 17), *qdnw = g(33, 18), *qdnb = g(32, 19);
  const float* ctxw = g(6, 23), *ctxdnw = g(5, 24), *ctxdnb = g(4, 25);
  const float* phiw = g(29, 26);
  const float* prew = g(31, 27), *preb = g(30, 28);
  const float* zw = g(14, 29), *zb = g(13, 30);
  const float* rw = g(12, 31), *rb = g(11, 32);
  const float* qw = g(10, 33), *qb = g(9, 34);
  const float* logw = g(28, 35), *logb = g(27, 36);
  const float* dltw = g(8, 37),  *dltb = g(7, 38);

  // ---------------- workspace arena (bf16 channel-planes of P_ pixels) ------
  unsigned short* arena = (unsigned short*)d_ws;
  auto plane = [&](int q) { return arena + (size_t)q * P_; };
  unsigned short* I32  = plane(0);     // 32ch: padded image / resized Fq
  unsigned short* T1   = plane(32);    // 64ch
  unsigned short* T2   = plane(96);    // 64ch (Fh_L)
  unsigned short* T3   = plane(160);   // 64ch (ctx)
  unsigned short* T4   = plane(224);   // 64ch (Fh_R)
  unsigned short* C128 = plane(288);   // 128ch ctx concat
  unsigned short* FLb  = plane(288);   // 64ch (aliases C128, after ctx conv)
  unsigned short* FRb  = plane(352);   // 64ch
  unsigned short* G96  = plane(416);   // 96ch cost|ctx concat
  unsigned short* Xb   = plane(32);    // 96ch (aliases T1/T2 region)
  unsigned short* Hb   = plane(160);   // 96ch (aliases T3/T4 region)
  unsigned short* XH   = plane(256);   // 192ch (aliases T4 tail + C128 + G96 head)
  unsigned short* ZR   = plane(448);   // 192ch
  unsigned short* Qb   = plane(640);   // 96ch
  unsigned short* LD   = plane(0);     // 10ch logits|delta (aliases I32)

  unsigned short* wcur = plane(736);
  auto alloc_w = [&](size_t el) { unsigned short* r = wcur; wcur += el; return r; };
  auto wsz = [](int O, int CINP, int KS) {
    return (size_t)(KS * KS * CINP / 32) * (size_t)((O + 15) / 16) * 512;
  };
  unsigned short* wp0    = alloc_w(wsz(64, 32, 3));
  unsigned short* wp1    = alloc_w(wsz(64, 64, 3));
  unsigned short* wp2    = alloc_w(wsz(64, 64, 3));
  unsigned short* wp3    = alloc_w(wsz(64, 64, 3));
  unsigned short* wpproj = alloc_w(wsz(64, 32, 1));
  unsigned short* wpctx  = alloc_w(wsz(64, 128, 3));
  unsigned short* wpphi  = alloc_w(wsz(64, 64, 1));
  unsigned short* wppre  = alloc_w(wsz(96, 96, 3));
  unsigned short* wpzr   = alloc_w(wsz(192, 192, 3));
  unsigned short* wpq    = alloc_w(wsz(96, 192, 3));
  unsigned short* wpld   = alloc_w(wsz(10, 96, 3));
  float* stats = (float*)wcur;  // 128 floats

  // ---------------- weight packing -----------------------------------------
  auto pack = [&](const float* w1, int O1, const float* w2, int O2,
                  int I, int CINP, int KS, unsigned short* dst) {
    size_t total = (size_t)(KS * KS * CINP / 32) * (size_t)(((O1 + O2) + 15) / 16) * 512;
    pack_weights<<<blocks1d(total), 256, 0, stream>>>(w1, O1, w2, O2, I, CINP, KS, dst);
  };
  pack(conv0w, 64, nullptr, 0, 3, 32, 3, wp0);
  pack(conv1w, 64, nullptr, 0, 64, 64, 3, wp1);
  pack(conv2w, 64, nullptr, 0, 64, 64, 3, wp2);
  pack(conv3w, 64, nullptr, 0, 64, 64, 3, wp3);
  pack(qprojw, 64, nullptr, 0, 32, 32, 1, wpproj);
  pack(ctxw, 64, nullptr, 0, 128, 128, 3, wpctx);
  pack(phiw, 64, nullptr, 0, 64, 64, 1, wpphi);
  pack(prew, 96, nullptr, 0, 73, 96, 3, wppre);
  pack(zw, 96, rw, 96, 192, 192, 3, wpzr);
  pack(qw, 96, nullptr, 0, 192, 192, 3, wpq);
  pack(logw, 9, dltw, 1, 96, 96, 3, wpld);

  auto dn = [&](unsigned short* buf, const float* w, const float* b) {
    zero_f32<<<1, 128, 0, stream>>>(stats, 128);
    chan_stats64<<<512, 256, 0, stream>>>(buf, stats);
    dnorm_silu64<<<blocks1d(P_), 256, 0, stream>>>(buf, stats, w, b);
  };

  // ---------------- hr encoders --------------------------------------------
  pad_image32<<<blocks1d((size_t)P_ * 32), 256, 0, stream>>>(left, I32);
  CONV(32, 64, 3, 2, 2, I32, wp0, nullptr, nullptr, 0, T1, 1, 0); dn(T1, dn0w, dn0b);
  CONV(64, 64, 3, 2, 2, T1, wp1, nullptr, nullptr, 0, T2, 2, 0);  dn(T2, dn1w, dn1b);
  CONV(64, 64, 3, 2, 2, T2, wp2, nullptr, nullptr, 0, T1, 4, 0);  dn(T1, dn2w, dn2b);
  CONV(64, 64, 3, 2, 2, T1, wp3, nullptr, nullptr, 0, T2, 8, 0);  dn(T2, dn3w, dn3b); // Fh_L

  pad_image32<<<blocks1d((size_t)P_ * 32), 256, 0, stream>>>(right, I32);
  CONV(32, 64, 3, 2, 2, I32, wp0, nullptr, nullptr, 0, T3, 1, 0); dn(T3, dn0w, dn0b);
  CONV(64, 64, 3, 2, 2, T3, wp1, nullptr, nullptr, 0, T4, 2, 0);  dn(T4, dn1w, dn1b);
  CONV(64, 64, 3, 2, 2, T4, wp2, nullptr, nullptr, 0, T3, 4, 0);  dn(T3, dn2w, dn2b);
  CONV(64, 64, 3, 2, 2, T3, wp3, nullptr, nullptr, 0, T4, 8, 0);  dn(T4, dn3w, dn3b); // Fh_R

  // ---------------- quarter adapt (left only; right branch is dead code) ---
  resize_fq<<<blocks1d((size_t)P_ * 32), 256, 0, stream>>>(fqL, I32);
  CONV(32, 64, 1, 2, 2, I32, wpproj, nullptr, nullptr, 0, T1, 1, 0);
  dn(T1, qdnw, qdnb);                                                  // FqL_up

  // ---------------- context -------------------------------------------------
  copy_ch<<<blocks1d((size_t)P_ * 64), 256, 0, stream>>>(T2, 64, C128, 128, 0, 64);
  copy_ch<<<blocks1d((size_t)P_ * 64), 256, 0, stream>>>(T1, 64, C128, 128, 64, 64);
  CONV(128, 64, 3, 2, 2, C128, wpctx, nullptr, nullptr, 0, T3, 1, 0);
  dn(T3, ctxdnw, ctxdnb);                                              // ctx

  // ---------------- phi embeddings + correlation ---------------------------
  CONV(64, 64, 1, 2, 2, T2, wpphi, nullptr, nullptr, 0, FLb, 1, 0);
  l2norm64<<<blocks1d(P_), 256, 0, stream>>>(FLb);
  CONV(64, 64, 1, 2, 2, T4, wpphi, nullptr, nullptr, 0, FRb, 1, 0);
  l2norm64<<<blocks1d(P_), 256, 0, stream>>>(FRb);

  zero_ch<<<blocks1d((size_t)P_ * 23), 256, 0, stream>>>(G96, 96, 73, 23); // K pad
  local_corr9<<<blocks1d(P_), 256, 0, stream>>>(FLb, FRb, d0, G96);
  copy_ch<<<blocks1d((size_t)P_ * 64), 256, 0, stream>>>(T3, 64, G96, 96, 9, 64);

  // ---------------- pre conv + GRU -----------------------------------------
  CONV(96, 96, 3, 2, 2, G96, wppre, preb, nullptr, 96, Xb, 1, 2);      // SiLU
  zero_ch<<<blocks1d((size_t)P_ * 96), 256, 0, stream>>>(Hb, 96, 0, 96);
  copy_ch<<<blocks1d((size_t)P_ * 96), 256, 0, stream>>>(Xb, 96, XH, 192, 0, 96);
  for (int it = 0; it < 2; ++it) {
    copy_ch<<<blocks1d((size_t)P_ * 96), 256, 0, stream>>>(Hb, 96, XH, 192, 96, 96);
    CONV(192, 192, 3, 2, 2, XH, wpzr, zb, rb, 96, ZR, 1, 3);           // sigmoid
    ew_rh<<<blocks1d((size_t)P_ * 96), 256, 0, stream>>>(ZR, Hb, XH);
    CONV(192, 96, 3, 2, 2, XH, wpq, qb, nullptr, 96, Qb, 1, 4);        // tanh
    ew_hupd<<<blocks1d((size_t)P_ * 96), 256, 0, stream>>>(ZR, Qb, Hb);
  }

  // ---------------- heads + output -----------------------------------------
  CONV(96, 10, 3, 1, 2, Hb, wpld, logb, dltb, 9, LD, 1, 1);            // bias only
  final_out<<<blocks1d(P_), 256, 0, stream>>>(LD, d0, (float*)d_out);
}
